// DGCNN_28415503631073
// MI455X (gfx1250) — matrix operand order repaired and checked by reading, not compiled, not run on hardware
//
#include <hip/hip_runtime.h>

#define KNN   20
#define NPTS  4096
#define BATCH 4

typedef _Float16 half_t;
typedef __attribute__((ext_vector_type(16))) _Float16 v16h;
typedef __attribute__((ext_vector_type(8)))  float    v8f;
typedef __attribute__((ext_vector_type(4)))  float    f4;
typedef __attribute__((ext_vector_type(4)))  unsigned u4;

union AFrag { v16h v; half_t h[16]; };
union CFrag { v8f  v; float  f[8];  };
union F16   { f4   q[4]; float f[16]; };
union H8    { u4   u; half_t h[8];   };

__device__ __forceinline__ f4 ldf4(const float* p) { return *(const f4*)p; }

__device__ __forceinline__ unsigned f2ord(float f) {
  unsigned b = __float_as_uint(f);
  return b ^ ((b & 0x80000000u) ? 0xFFFFFFFFu : 0x80000000u);
}
__device__ __forceinline__ float ord2f(unsigned u) {
  unsigned b = u ^ ((u & 0x80000000u) ? 0x80000000u : 0xFFFFFFFFu);
  return __uint_as_float(b);
}
__device__ __forceinline__ float lrelu(float x) { return x > 0.f ? x : 0.2f * x; }
__device__ __forceinline__ float bnscale(float g) { return g * rsqrtf(1.f + 1e-5f); }

// ---------------------------------------------------------------------------
// f32 -> f16 pre-conversion (weights)
// ---------------------------------------------------------------------------
__global__ void cvt_f16_kernel(const float* __restrict__ src,
                               half_t* __restrict__ dst, int n) {
  int t = blockIdx.x * blockDim.x + threadIdx.x;
  if (t < n) dst[t] = (half_t)src[t];
}

// ---------------------------------------------------------------------------
// points (B,3,N) -> pts_t (B*N, 3)
// ---------------------------------------------------------------------------
__global__ void transpose_pts_kernel(const float* __restrict__ pts,
                                     float* __restrict__ ptst) {
  int t = blockIdx.x * blockDim.x + threadIdx.x;
  if (t >= BATCH * NPTS) return;
  int b = t >> 12, n = t & (NPTS - 1);
  #pragma unroll
  for (int c = 0; c < 3; ++c)
    ptst[(size_t)t * 3 + c] = pts[((size_t)b * 3 + c) * NPTS + n];
}

// ---------------------------------------------------------------------------
// per-point squared norms. X laid out (B*N, stride), first C channels used.
// ---------------------------------------------------------------------------
__global__ void sqnorm_kernel(const float* __restrict__ X, int stride, int C,
                              float* __restrict__ sq) {
  int t = blockIdx.x * blockDim.x + threadIdx.x;
  if (t >= BATCH * NPTS) return;
  const float* p = X + (size_t)t * stride;
  float s = 0.f;
  for (int c = 0; c < C; ++c) s += p[c] * p[c];
  sq[t] = s;
}

// ---------------------------------------------------------------------------
// Fused knn: one wave per row. fp32 distances (ordering-critical), per-lane
// sorted top-20 in registers, wave-wide argmax merge via shfl_xor (wave32).
// ---------------------------------------------------------------------------
template <int C>
__global__ void __launch_bounds__(256)
knn_kernel(const float* __restrict__ X, int stride,
           const float* __restrict__ sq, int* __restrict__ knn_out) {
  constexpr int NV = (C + 3) / 4;
  int wave = threadIdx.x >> 5;
  int lane = threadIdx.x & 31;
  int row  = blockIdx.x * 8 + wave;          // 0 .. B*N-1
  int b    = row >> 12;

  const float* rp = X + (size_t)row * stride;
  f4 rv4[NV];
  if (C % 4 == 0) {
    #pragma unroll
    for (int q = 0; q < NV; ++q) rv4[q] = ldf4(rp + 4 * q);
  } else {                                   // C==3
    rv4[0][0] = rp[0]; rv4[0][1] = rp[1]; rv4[0][2] = rp[2]; rv4[0][3] = 0.f;
  }
  float sqi = sq[row];

  float lv[KNN]; int li[KNN];
  #pragma unroll
  for (int t = 0; t < KNN; ++t) { lv[t] = -3.0e38f; li[t] = 0x7fffffff; }

  for (int j = lane; j < NPTS; j += 32) {
    const float* cp = X + (size_t)(b * NPTS + j) * stride;
    float dot;
    if (C % 4 == 0) {
      f4 a = rv4[0] * ldf4(cp);
      #pragma unroll
      for (int q = 1; q < NV; ++q) a += rv4[q] * ldf4(cp + 4 * q);
      dot = a[0] + a[1] + a[2] + a[3];
    } else {
      dot = rv4[0][0] * cp[0] + rv4[0][1] * cp[1] + rv4[0][2] * cp[2];
    }
    float nd = 2.f * dot - sqi - sq[b * NPTS + j];
    // unrolled sorted-descending insertion (stable: strict > keeps earlier j)
    #pragma unroll
    for (int t = KNN - 1; t >= 1; --t) {
      bool shift = nd > lv[t - 1];
      bool place = (nd > lv[t]) && !shift;
      if (shift)      { lv[t] = lv[t - 1]; li[t] = li[t - 1]; }
      else if (place) { lv[t] = nd;        li[t] = j;         }
    }
    if (nd > lv[0]) { lv[0] = nd; li[0] = j; }
  }

  // 20-round wave argmax merge (ties -> smaller index, matching lax.top_k)
  int* op = knn_out + (size_t)row * KNN;
  for (int r = 0; r < KNN; ++r) {
    float bv = lv[0]; int bi = li[0];
    #pragma unroll
    for (int off = 16; off >= 1; off >>= 1) {
      float ov = __shfl_xor(bv, off, 32);
      int   oi = __shfl_xor(bi, off, 32);
      if (ov > bv || (ov == bv && oi < bi)) { bv = ov; bi = oi; }
    }
    if (lane == 0) op[r] = bi;
    if (lv[0] == bv && li[0] == bi) {          // this lane won: pop head
      #pragma unroll
      for (int t = 0; t < KNN - 1; ++t) { lv[t] = lv[t + 1]; li[t] = li[t + 1]; }
      lv[KNN - 1] = -3.0e38f; li[KNN - 1] = 0x7fffffff;
    }
  }
}

// ---------------------------------------------------------------------------
// Fragment loaders. WMMA f16 K-maps used here (contiguous, not ISA table order):
//   A lane element e -> K = ks + e + hi*16                 (one 16-elem run)
//   B lane element e -> K = ks + e + hi*16                 (one 16-elem run)
// ---------------------------------------------------------------------------
__device__ __forceinline__ void loadA16h(const half_t* __restrict__ wrow,
                                         int ka0, AFrag& afr) {
  H8 r0, r1;
  r0.u = *(const u4*)(wrow + ka0);           // 8 halves, 16B
  r1.u = *(const u4*)(wrow + ka0 + 8);       // 8 halves, 16B
  #pragma unroll
  for (int e = 0; e < 8; ++e) { afr.h[e] = r0.h[e]; afr.h[8 + e] = r1.h[e]; }
}
__device__ __forceinline__ void loadB16h(const half_t* __restrict__ bp,
                                         AFrag& bfr) {
  H8 r0, r1;
  r0.u = *(const u4*)bp;                     // 16 contiguous halves, 32B
  r1.u = *(const u4*)(bp + 8);
  #pragma unroll
  for (int e = 0; e < 8; ++e) { bfr.h[e] = r0.h[e]; bfr.h[8 + e] = r1.h[e]; }
}
__device__ __forceinline__ void loadB16(const float* __restrict__ bp,
                                        AFrag& bfr) {
  F16 bx;
  #pragma unroll
  for (int q = 0; q < 4; ++q) bx.q[q] = ldf4(bp + 4 * q);
  #pragma unroll
  for (int e = 0; e < 16; ++e) bfr.h[e] = (half_t)bx.f[e];
}
__device__ __forceinline__ void loadB16diff(const float* __restrict__ np,
                                            const float* __restrict__ cp,
                                            AFrag& bfr) {
  F16 bx;
  #pragma unroll
  for (int q = 0; q < 4; ++q) bx.q[q] = ldf4(np + 4 * q) - ldf4(cp + 4 * q);
  #pragma unroll
  for (int e = 0; e < 16; ++e) bfr.h[e] = (half_t)bx.f[e];
}

// ---------------------------------------------------------------------------
// Fused edge-conv block: gather -> convA(+bn+lrelu) [-> convB(+bn+lrelu)]
// -> max over k. One workgroup = 16 points = 320 (n,k) columns.
// 4 waves; wave w owns output channels [16w,16w+16). WMMA f16->f32.
// A-fragments (weights) hoisted into registers across all 20 column tiles.
// Emits both f32 (for knn/gather) and f16 (for downstream GEMMs) outputs.
// ---------------------------------------------------------------------------
template <int CIN, bool TWO>
__global__ void __launch_bounds__(128)
edge_conv_kernel(const float* __restrict__ X, int stride,
                 const int* __restrict__ knn_in,
                 const half_t* __restrict__ Wa, const float* __restrict__ ba,
                 const float* __restrict__ ga, const float* __restrict__ bea,
                 const half_t* __restrict__ Wb, const float* __restrict__ bbv,
                 const float* __restrict__ gb, const float* __restrict__ beb,
                 float* __restrict__ out, half_t* __restrict__ outH,
                 int outStride) {
  constexpr int KA  = (2 * CIN + 31) & ~31;    // padded K for convA
  constexpr int NKS = KA / 32;
  __shared__ half_t   y1L[320 * 64];           // convA output (f16), 40 KB
  __shared__ unsigned maxL[16 * 64];           // per-point per-channel max keys

  int tid  = threadIdx.x;
  int wave = tid >> 5, lane = tid & 31;
  int lo   = lane & 15, hi = lane >> 4;
  int b    = blockIdx.x >> 8;
  int p0   = (blockIdx.x & 255) * 16;

  for (int t = tid; t < 16 * 64; t += 128) maxL[t] = 0u;
  __syncthreads();

  int   ch0 = wave * 16 + hi * 8;              // 8 contiguous output channels
  F16   bav, gav, beav;
  bav.q[0]  = ldf4(ba  + ch0); bav.q[1]  = ldf4(ba  + ch0 + 4);
  gav.q[0]  = ldf4(ga  + ch0); gav.q[1]  = ldf4(ga  + ch0 + 4);
  beav.q[0] = ldf4(bea + ch0); beav.q[1] = ldf4(bea + ch0 + 4);

  // hoisted convA A-fragments (weight rows are ct-invariant)
  AFrag afrA[NKS];
  if (CIN % 16 == 0) {
    #pragma unroll
    for (int s = 0; s < NKS; ++s)
      loadA16h(Wa + (size_t)(wave * 16 + lo) * (2 * CIN), s * 32 + hi * 16,
               afrA[s]);
  } else {                                     // CIN==3: rows of 6, scalar
    #pragma unroll
    for (int e = 0; e < 16; ++e) {
      int ka = e + hi * 16;
      int m  = wave * 16 + lo;
      afrA[0].h[e] = (ka < 2 * CIN) ? Wa[(size_t)m * (2 * CIN) + ka]
                                    : (half_t)0.f;
    }
  }

  // ---- stage A: convA over gathered edge features --------------------------
  for (int ct = 0; ct < 20; ++ct) {
    int c   = ct * 16 + lo;                    // column 0..319
    int pl  = c / 20;                          // point-local 0..15
    int kk  = c % 20;
    int p   = p0 + pl;
    int nbr = knn_in[((size_t)b * NPTS + p) * KNN + kk];
    const float* ctrp = X + (size_t)(b * NPTS + p)   * stride;
    const float* nbrp = X + (size_t)(b * NPTS + nbr) * stride;

    CFrag acc;
    #pragma unroll
    for (int r = 0; r < 8; ++r) acc.f[r] = 0.f;

    #pragma unroll
    for (int s = 0; s < NKS; ++s) {
      AFrag bfr;
      int kb0 = s * 32 + hi * 16;
      if (CIN % 16 == 0) {                     // CIN==64: vector gather path
        if (kb0 < CIN) loadB16diff(nbrp + kb0, ctrp + kb0, bfr);
        else           loadB16(ctrp + (kb0 - CIN), bfr);
      } else {                                 // CIN==3: tiny scalar path
        #pragma unroll
        for (int e = 0; e < 16; ++e) {
          int kb = kb0 + e;
          float bv = 0.f;
          if      (kb < CIN)     bv = nbrp[kb] - ctrp[kb];
          else if (kb < 2 * CIN) bv = ctrp[kb - CIN];
          bfr.h[e] = (half_t)bv;
        }
      }
      acc.v = __builtin_amdgcn_wmma_f32_16x16x32_f16(
          false, afrA[s].v, false, bfr.v, (short)0, acc.v, false, false);
    }
    if (TWO) {
      H8 st;
      #pragma unroll
      for (int r = 0; r < 8; ++r) {
        float y = (acc.f[r] + bav.f[r]) * bnscale(gav.f[r]) + beav.f[r];
        st.h[r] = (half_t)lrelu(y);
      }
      *(u4*)&y1L[c * 64 + ch0] = st.u;         // one b128 DS store
    } else {
      #pragma unroll
      for (int r = 0; r < 8; ++r) {
        float y = (acc.f[r] + bav.f[r]) * bnscale(gav.f[r]) + beav.f[r];
        atomicMax(&maxL[pl * 64 + ch0 + r], f2ord(lrelu(y)));
      }
    }
  }

  // ---- stage B: convB from LDS, then max over k ----------------------------
  if (TWO) {
    F16 bbvv, gbv, bebv;
    bbvv.q[0] = ldf4(bbv + ch0); bbvv.q[1] = ldf4(bbv + ch0 + 4);
    gbv.q[0]  = ldf4(gb  + ch0); gbv.q[1]  = ldf4(gb  + ch0 + 4);
    bebv.q[0] = ldf4(beb + ch0); bebv.q[1] = ldf4(beb + ch0 + 4);
    AFrag afrB[2];
    #pragma unroll
    for (int s = 0; s < 2; ++s)
      loadA16h(Wb + (size_t)(wave * 16 + lo) * 64, s * 32 + hi * 16, afrB[s]);
    __syncthreads();
    for (int ct = 0; ct < 20; ++ct) {
      int c  = ct * 16 + lo;
      int pl = c / 20;
      CFrag acc;
      #pragma unroll
      for (int r = 0; r < 8; ++r) acc.f[r] = 0.f;
      #pragma unroll
      for (int s = 0; s < 2; ++s) {
        AFrag bfr;
        const half_t* yp = &y1L[c * 64 + s * 32 + hi * 16];
        H8 r0, r1;
        r0.u = *(const u4*)yp;                 // two b128 DS loads
        r1.u = *(const u4*)(yp + 8);
        #pragma unroll
        for (int e = 0; e < 8; ++e) { bfr.h[e] = r0.h[e]; bfr.h[8 + e] = r1.h[e]; }
        acc.v = __builtin_amdgcn_wmma_f32_16x16x32_f16(
            false, afrB[s].v, false, bfr.v, (short)0, acc.v, false, false);
      }
      #pragma unroll
      for (int r = 0; r < 8; ++r) {
        float y = (acc.f[r] + bbvv.f[r]) * bnscale(gbv.f[r]) + bebv.f[r];
        atomicMax(&maxL[pl * 64 + ch0 + r], f2ord(lrelu(y)));
      }
    }
  }

  __syncthreads();
  // writeback: 128 threads x 8 contiguous channels = 16 points x 64 channels
  {
    int pl = tid >> 3, cg = (tid & 7) * 8;
    F16 o; H8 oh;
    #pragma unroll
    for (int r = 0; r < 8; ++r) {
      float v = ord2f(maxL[pl * 64 + cg + r]);
      o.f[r] = v; oh.h[r] = (half_t)v;
    }
    size_t base = (size_t)(b * NPTS + p0 + pl) * outStride + cg;
    float* po = out + base;
    *(f4*)po = o.q[0]; *(f4*)(po + 4) = o.q[1];
    *(u4*)(outH + base) = oh.u;
  }
}

// ---------------------------------------------------------------------------
// Pointwise conv GEMM: one wave per 16x16 output tile, f16 weights AND f16
// activations (pre-stored). B = [Xg (per-batch, Cg ch)] ++ [X (per-col, Cx)].
// maxOut != null: fused global max over columns (conv6).
// outH != null: store f16 (col, Cout). Else outF: (b, ch, n) f32 (final).
// ---------------------------------------------------------------------------
__global__ void __launch_bounds__(32)
pw_conv_kernel(const half_t* __restrict__ W, const float* __restrict__ bias,
               const float* __restrict__ gamma, const float* __restrict__ beta,
               const half_t* __restrict__ Xg, int Cg,
               const half_t* __restrict__ X,  int Cx,
               int Cout, half_t* __restrict__ outH, float* __restrict__ outF,
               unsigned* __restrict__ maxOut, int hasBN) {
  int lane = threadIdx.x & 31;
  int lo = lane & 15, hi = lane >> 4;
  int ct = blockIdx.x, mt = blockIdx.y;
  int col = ct * 16 + lo;
  int b   = col >> 12;
  int Kt  = Cg + Cx;
  const half_t* xcol = X + (size_t)col * Cx;
  const half_t* xgp  = Xg ? (Xg + (size_t)b * Cg) : nullptr;
  int m = mt * 16 + lo;
  const half_t* wrow = W + (size_t)m * Kt;   // W padded: rows always valid

  CFrag acc;
  #pragma unroll
  for (int r = 0; r < 8; ++r) acc.f[r] = 0.f;

  for (int ks = 0; ks < Kt; ks += 32) {
    AFrag afr, bfr;
    loadA16h(wrow, ks + hi * 16, afr);
    int kb0 = ks + hi * 16;                    // run never straddles Cg
    const half_t* bp = (kb0 < Cg) ? (xgp + kb0) : (xcol + (kb0 - Cg));
    loadB16h(bp, bfr);
    acc.v = __builtin_amdgcn_wmma_f32_16x16x32_f16(
        false, afr.v, false, bfr.v, (short)0, acc.v, false, false);
  }

  int ch0 = mt * 16 + hi * 8;
  if (maxOut) {  // conv6: bn+lrelu then max over the 16 columns, atomic to xg
    F16 bv_, gv_, bev_;
    bv_.q[0]  = ldf4(bias  + ch0); bv_.q[1]  = ldf4(bias  + ch0 + 4);
    gv_.q[0]  = ldf4(gamma + ch0); gv_.q[1]  = ldf4(gamma + ch0 + 4);
    bev_.q[0] = ldf4(beta  + ch0); bev_.q[1] = ldf4(beta  + ch0 + 4);
    #pragma unroll
    for (int r = 0; r < 8; ++r) {
      float y = (acc.f[r] + bv_.f[r]) * bnscale(gv_.f[r]) + bev_.f[r];
      y = lrelu(y);
      #pragma unroll
      for (int off = 8; off >= 1; off >>= 1)
        y = fmaxf(y, __shfl_xor(y, off, 32));
      if (lo == 0) atomicMax(&maxOut[(size_t)b * Cout + ch0 + r], f2ord(y));
    }
  } else if (outH) {                           // f16 store, Cout mult of 16
    F16 bv_, gv_, bev_;
    H8 o;
    bv_.q[0] = ldf4(bias + ch0); bv_.q[1] = ldf4(bias + ch0 + 4);
    if (hasBN) {
      gv_.q[0]  = ldf4(gamma + ch0); gv_.q[1]  = ldf4(gamma + ch0 + 4);
      bev_.q[0] = ldf4(beta  + ch0); bev_.q[1] = ldf4(beta  + ch0 + 4);
    }
    #pragma unroll
    for (int r = 0; r < 8; ++r) {
      float y = acc.f[r] + bv_.f[r];
      if (hasBN) { y = y * bnscale(gv_.f[r]) + bev_.f[r]; y = lrelu(y); }
      o.h[r] = (half_t)y;
    }
    *(u4*)(outH + (size_t)col * Cout + ch0) = o.u;
  } else {                                     // final: (b, ch, n) f32, Cout=13
    #pragma unroll
    for (int r = 0; r < 8; ++r) {
      int ch = ch0 + r;
      if (ch >= Cout) continue;
      float y = acc.f[r] + bias[ch];
      if (hasBN) { y = y * bnscale(gamma[ch]) + beta[ch]; y = lrelu(y); }
      outF[((size_t)b * Cout + ch) * NPTS + (col & (NPTS - 1))] = y;
    }
  }
}

__global__ void init_u32_kernel(unsigned* __restrict__ p, int n) {
  int t = blockIdx.x * blockDim.x + threadIdx.x;
  if (t < n) p[t] = 0u;
}
__global__ void ord2h_kernel(const unsigned* __restrict__ u,
                             half_t* __restrict__ h, int n) {
  int t = blockIdx.x * blockDim.x + threadIdx.x;
  if (t < n) h[t] = (half_t)ord2f(u[t]);
}

// ---------------------------------------------------------------------------
extern "C" void kernel_launch(void* const* d_in, const int* in_sizes, int n_in,
                              void* d_out, int out_size, void* d_ws,
                              size_t ws_size, hipStream_t stream) {
  (void)in_sizes; (void)n_in; (void)out_size;
  const float* points = (const float*)d_in[0];
  const float *w1 = (const float*)d_in[1],  *b1 = (const float*)d_in[2];
  const float *w2 = (const float*)d_in[3],  *b2 = (const float*)d_in[4];
  const float *w3 = (const float*)d_in[5],  *b3 = (const float*)d_in[6];
  const float *w4 = (const float*)d_in[7],  *b4 = (const float*)d_in[8];
  const float *w5 = (const float*)d_in[9],  *b5 = (const float*)d_in[10];
  const float *w6 = (const float*)d_in[11], *b6 = (const float*)d_in[12];
  const float *w7 = (const float*)d_in[13], *b7 = (const float*)d_in[14];
  const float *w8 = (const float*)d_in[15], *b8 = (const float*)d_in[16];
  const float *w9 = (const float*)d_in[17], *b9 = (const float*)d_in[18];
  const float *g1 = (const float*)d_in[19], *be1 = (const float*)d_in[20];
  const float *g2 = (const float*)d_in[21], *be2 = (const float*)d_in[22];
  const float *g3 = (const float*)d_in[23], *be3 = (const float*)d_in[24];
  const float *g4 = (const float*)d_in[25], *be4 = (const float*)d_in[26];
  const float *g5 = (const float*)d_in[27], *be5 = (const float*)d_in[28];
  const float *g6 = (const float*)d_in[29], *be6 = (const float*)d_in[30];
  const float *g7 = (const float*)d_in[31], *be7 = (const float*)d_in[32];
  const float *g8 = (const float*)d_in[33], *be8 = (const float*)d_in[34];

  char* ws = (char*)d_ws;
  size_t off = 0;
  auto alloc = [&](size_t bytes) -> char* {
    char* p = ws + off;
    off = (off + bytes + 255) & ~(size_t)255;
    return p;
  };
  const size_t BN = (size_t)BATCH * NPTS;
  float*    pts_t  = (float*)   alloc(BN * 3 * 4);
  float*    sq     = (float*)   alloc(BN * 4);
  int*      knnix  = (int*)     alloc(BN * KNN * 4);
  float*    xcat   = (float*)   alloc(BN * 192 * 4);     // x1|x2|x3 (f32)
  half_t*   xcat_h = (half_t*)  alloc(BN * 192 * 2);     // x1|x2|x3 (f16)
  unsigned* xg_u   = (unsigned*)alloc((size_t)BATCH * 1024 * 4);
  half_t*   xg_h   = (half_t*)  alloc((size_t)BATCH * 1024 * 2);
  half_t*   y7_h   = (half_t*)  alloc(BN * 512 * 2);
  half_t*   y8_h   = (half_t*)  alloc(BN * 256 * 2);
  // f16 weight copies; w9 padded to 16 rows so A loads need no guard
  const int wsz[9]  = {64 * 6, 64 * 64, 64 * 128, 64 * 64, 64 * 128,
                       1024 * 192, 512 * 1216, 256 * 512, 13 * 256};
  const int wpad[9] = {64 * 6, 64 * 64, 64 * 128, 64 * 64, 64 * 128,
                       1024 * 192, 512 * 1216, 256 * 512, 16 * 256};
  half_t* wh[9];
  for (int i = 0; i < 9; ++i) wh[i] = (half_t*)alloc((size_t)wpad[i] * 2);
  if (off > ws_size) return;   // insufficient workspace

  // zero w9 pad rows, then convert all weights to f16
  init_u32_kernel<<<(16 * 256 / 2 + 255) / 256, 256, 0, stream>>>(
      (unsigned*)wh[8], 16 * 256 / 2);
  const float* wsrc[9] = {w1, w2, w3, w4, w5, w6, w7, w8, w9};
  for (int i = 0; i < 9; ++i)
    cvt_f16_kernel<<<(wsz[i] + 255) / 256, 256, 0, stream>>>(wsrc[i], wh[i],
                                                             wsz[i]);

  float* out = (float*)d_out;
  dim3 thr256(256), thr128(128), thr32(32);

  // points -> (B*N,3)
  transpose_pts_kernel<<<64, thr256, 0, stream>>>(points, pts_t);

  // ---- block 1: knn on coords (C=3), edge conv w1->w2, max -> x1 ----------
  sqnorm_kernel<<<64, thr256, 0, stream>>>(pts_t, 3, 3, sq);
  knn_kernel<3><<<2048, thr256, 0, stream>>>(pts_t, 3, sq, knnix);
  edge_conv_kernel<3, true><<<1024, thr128, 0, stream>>>(
      pts_t, 3, knnix, wh[0], b1, g1, be1, wh[1], b2, g2, be2,
      xcat + 0, xcat_h + 0, 192);

  // ---- block 2: knn on x1 (C=64), edge conv w3->w4, max -> x2 -------------
  sqnorm_kernel<<<64, thr256, 0, stream>>>(xcat + 0, 192, 64, sq);
  knn_kernel<64><<<2048, thr256, 0, stream>>>(xcat + 0, 192, sq, knnix);
  edge_conv_kernel<64, true><<<1024, thr128, 0, stream>>>(
      xcat + 0, 192, knnix, wh[2], b3, g3, be3, wh[3], b4, g4, be4,
      xcat + 64, xcat_h + 64, 192);

  // ---- block 3: knn on x2 (C=64), edge conv w5 only, max -> x3 ------------
  sqnorm_kernel<<<64, thr256, 0, stream>>>(xcat + 64, 192, 64, sq);
  knn_kernel<64><<<2048, thr256, 0, stream>>>(xcat + 64, 192, sq, knnix);
  edge_conv_kernel<64, false><<<1024, thr128, 0, stream>>>(
      xcat + 64, 192, knnix, wh[4], b5, g5, be5,
      nullptr, nullptr, nullptr, nullptr, xcat + 128, xcat_h + 128, 192);

  // ---- conv6 (1024x192) fused with global max over n -> xg ----------------
  init_u32_kernel<<<16, thr256, 0, stream>>>(xg_u, BATCH * 1024);
  pw_conv_kernel<<<dim3(1024, 64), thr32, 0, stream>>>(
      wh[5], b6, g6, be6, nullptr, 0, xcat_h, 192, 1024,
      nullptr, nullptr, xg_u, 1);
  ord2h_kernel<<<16, thr256, 0, stream>>>(xg_u, xg_h, BATCH * 1024);

  // ---- conv7 (512x1216) on [xg | x1 x2 x3] -> y7 (f16) --------------------
  pw_conv_kernel<<<dim3(1024, 32), thr32, 0, stream>>>(
      wh[6], b7, g7, be7, xg_h, 1024, xcat_h, 192, 512,
      y7_h, nullptr, nullptr, 1);

  // ---- conv8 (256x512) -> y8 (f16) ----------------------------------------
  pw_conv_kernel<<<dim3(1024, 16), thr32, 0, stream>>>(
      wh[7], b8, g8, be8, nullptr, 0, y7_h, 512, 256,
      y8_h, nullptr, nullptr, 1);

  // ---- conv9 (13x256, padded to 16), bias only, output (B,13,N) -----------
  pw_conv_kernel<<<dim3(1024, 1), thr32, 0, stream>>>(
      wh[8], b9, nullptr, nullptr, nullptr, 0, y8_h, 256, 13,
      nullptr, out, nullptr, 0);
}
